// NCGCN_73804718014858
// MI455X (gfx1250) — compile-verified, matrix operand order checked by reading, not run on hardware
//
#include <hip/hip_runtime.h>

#define NN 100000
#define EE 1600000
#define FF 128
#define HH 128
#define CCL 10

typedef __attribute__((ext_vector_type(8)))  __bf16 v8bf;
typedef __attribute__((ext_vector_type(16))) __bf16 v16bf;
typedef __attribute__((ext_vector_type(8)))  float  v8f;

static __device__ __forceinline__ unsigned short f2bf_bits(float f) {
    union { float f; unsigned u; } in; in.f = f;
    return (unsigned short)((in.u + 0x7fffu + ((in.u >> 16) & 1u)) >> 16); // RNE
}

// ------------------------------------------------------------- conversion ---
__global__ void k_cvt_bf16(const float* __restrict__ src,
                           unsigned short* __restrict__ dst, int n) {
    int i = blockIdx.x * blockDim.x + threadIdx.x;
    if (i < n) dst[i] = f2bf_bits(src[i]);
}

// ---------------------------------------------------------------- degrees ---
__global__ void k_init_deg(float* __restrict__ deg) {
    int i = blockIdx.x * blockDim.x + threadIdx.x;
    if (i < 4 * NN) deg[i] = 1.0f;               // self-loop contribution
}

__global__ void k_edge_deg(const int* __restrict__ ei, const float* __restrict__ cc,
                           float* __restrict__ degL, float* __restrict__ degLL,
                           float* __restrict__ degH, float* __restrict__ degHH) {
    int e = blockIdx.x * blockDim.x + threadIdx.x;
    if (e >= EE) return;
    int col = ei[e];          // source
    int row = ei[EE + e];     // destination (segment target)
    if (row == col) return;   // diagonal removed before self-loop add
    float mr = cc[row], mc = cc[col];
    unsafeAtomicAdd(&degL [row], mr);
    unsafeAtomicAdd(&degLL[row], mc);
    unsafeAtomicAdd(&degH [row], 1.0f - mr);
    unsafeAtomicAdd(&degHH[row], 1.0f - mc);
}

__global__ void k_dinv(float* __restrict__ deg) {
    int i = blockIdx.x * blockDim.x + threadIdx.x;
    if (i < 4 * NN) deg[i] = rsqrtf(deg[i]);     // deg >= 1
}

__global__ void k_edge_vals(const int* __restrict__ ei, const float* __restrict__ cc,
                            const float* __restrict__ dL,  const float* __restrict__ dLL,
                            const float* __restrict__ dH,  const float* __restrict__ dHH,
                            float* __restrict__ vl, float* __restrict__ vll,
                            float* __restrict__ vh, float* __restrict__ vhh) {
    int e = blockIdx.x * blockDim.x + threadIdx.x;
    if (e >= EE) return;
    int col = ei[e], row = ei[EE + e];
    if (row == col) { vl[e] = 0.f; vll[e] = 0.f; vh[e] = 0.f; vhh[e] = 0.f; return; }
    float mr = cc[row], mc = cc[col];
    vl [e] = dL [row] * mr          * dL [col];
    vll[e] = dLL[row] * mc          * dLL[col];
    vh [e] = dH [row] * (1.f - mr)  * dH [col];
    vhh[e] = dHH[row] * (1.f - mc)  * dHH[col];
}

// ------------------------------------------------------------------- SpMM ---
// y[i,:] = dinv[i]^2 * x[i,:]  (explicit diagonal term, also zero-inits y)
__global__ void k_spmm_diag(const float* __restrict__ x, const float* __restrict__ dinv,
                            float* __restrict__ y) {
    size_t i = (size_t)blockIdx.x * blockDim.x + threadIdx.x;
    if (i >= (size_t)NN * FF) return;
    int node = (int)(i >> 7);
    float d = dinv[node];
    y[i] = d * d * x[i];
}

// y[row,:] += val[e] * x[col,:]   one wave32 per edge, 4 floats per lane
__global__ __launch_bounds__(256) void k_spmm_scatter(const int* __restrict__ ei,
                                                      const float* __restrict__ val,
                                                      const float* __restrict__ x,
                                                      float* __restrict__ y) {
    int e = blockIdx.x * 8 + (threadIdx.x >> 5);
    if (e >= EE) return;
    float v = val[e];
    if (v == 0.0f) return;                       // masked-out or diagonal edge
    int lane = threadIdx.x & 31;
    int col = ei[e];
    int row = ei[EE + e];
    const float* __restrict__ xs = x + (size_t)col * FF;
    float*       __restrict__ yd = y + (size_t)row * FF;
    #pragma unroll
    for (int i = 0; i < 4; ++i) {
        int f = lane + (i << 5);
        unsafeAtomicAdd(&yd[f], v * xs[f]);
    }
}

// ------------------------------------------------------- WMMA bf16 GEMM -----
// C[nrows, HOUT] = op(Abf[nrows,128] @ Wbf[128,HOUT] (+bias)).
// The whole weight matrix (<=128x128 bf16) is staged once per block in LDS
// (transposed [n][k], stride 136 elements = 272B = 17*16B for ds_load_b128
// alignment + bank spread). Each wave owns one 16-row tile, loads its A
// fragments once into registers (4 x v16bf), then sweeps all column tiles.
template<int HOUT, bool RELU, bool BIAS>
__global__ __launch_bounds__(256) void k_gemm(const unsigned short* __restrict__ A,
                                              const unsigned short* __restrict__ W,
                                              const float* __restrict__ bias,
                                              float* __restrict__ C, int nrows) {
    constexpr int NT = (HOUT + 15) / 16;         // column tiles
    __shared__ __align__(16) unsigned short ldsW[NT * 16 * 136];

    const int tid = threadIdx.x;
    for (int idx = tid; idx < NT * 16 * FF; idx += 256) {
        int n = idx >> 7, k = idx & 127;
        ldsW[n * 136 + k] = (n < HOUT) ? W[(size_t)k * HOUT + n]
                                       : (unsigned short)0;
    }
    __syncthreads();

    const int wave = tid >> 5;
    const int lane = tid & 31;
    const int half = lane >> 4;
    const int m    = lane & 15;
    const int row0 = (blockIdx.x * 8 + wave) * 16;
    if (row0 >= nrows) return;                   // wave-uniform, after barrier

    // A fragments: read each element of this row tile exactly once.
    const __bf16* __restrict__ arow = (const __bf16*)A + (size_t)(row0 + m) * FF;
    v16bf afrag[4];
    #pragma unroll
    for (int k0 = 0; k0 < FF; k0 += 32) {
        const int kb = k0 + (half << 3);
        v8bf alo = *(const v8bf*)(arow + kb);
        v8bf ahi = *(const v8bf*)(arow + kb + 16);
        afrag[k0 >> 5] = __builtin_shufflevector(alo, ahi, 0,1,2,3,4,5,6,7,
                                                           8,9,10,11,12,13,14,15);
    }

    #pragma unroll
    for (int t = 0; t < NT; ++t) {
        const int cn = t * 16 + m;
        const __bf16* __restrict__ brow = (const __bf16*)ldsW + (size_t)cn * 136;
        v8f acc = {};
        #pragma unroll
        for (int k0 = 0; k0 < FF; k0 += 32) {
            const int kb = k0 + (half << 3);
            v8bf blo = *(const v8bf*)(brow + kb);
            v8bf bhi = *(const v8bf*)(brow + kb + 16);
            v16bf b = __builtin_shufflevector(blo, bhi, 0,1,2,3,4,5,6,7,
                                                        8,9,10,11,12,13,14,15);
            acc = __builtin_amdgcn_wmma_f32_16x16x32_bf16(
                false, afrag[k0 >> 5], false, b, (short)0, acc, false, false);
        }
        if ((HOUT & 15) == 0 || cn < HOUT) {
            const float bv = BIAS ? bias[cn] : 0.0f;
            float* __restrict__ cptr =
                C + (size_t)(row0 + (half << 3)) * HOUT + cn;
            #pragma unroll
            for (int r = 0; r < 8; ++r) {        // VGPR r: M = r + half*8, N = m
                float v = acc[r] + bv;
                if (RELU) v = fmaxf(v, 0.0f);
                cptr[(size_t)r * HOUT] = v;
            }
        }
    }
}

// -------------------------------------------------------- gated fusion ------
// xf = relu(lamx*xc + laml*xl + lamh*xh), written directly as bf16 for the
// classifier GEMM.
__global__ void k_fuse(const float* __restrict__ xc, const float* __restrict__ xl,
                       const float* __restrict__ xh, const float* __restrict__ cc,
                       const float* __restrict__ lam1, const float* __restrict__ lam2,
                       unsigned short* __restrict__ xf) {
    size_t i = (size_t)blockIdx.x * blockDim.x + threadIdx.x;
    if (i >= (size_t)NN * HH) return;
    float a0 = lam1[0], a1 = lam1[1];
    float mA = fmaxf(a0, a1);
    float eA0 = __expf(a0 - mA), eA1 = __expf(a1 - mA);
    float invA = 1.0f / (eA0 + eA1);
    float lamxl = eA0 * invA, laml = eA1 * invA;
    float b0 = lam2[0], b1 = lam2[1];
    float mB = fmaxf(b0, b1);
    float eB0 = __expf(b0 - mB), eB1 = __expf(b1 - mB);
    float invB = 1.0f / (eB0 + eB1);
    float lamxh = eB0 * invB, lamh = eB1 * invB;

    int node = (int)(i >> 7);
    float c = cc[node];
    float lamx = lamxl * c + lamxh * (1.0f - c);
    float v = lamx * xc[i] + laml * xl[i] + lamh * xh[i];
    xf[i] = f2bf_bits(fmaxf(v, 0.0f));
}

// ------------------------------------------------------------------ driver --
extern "C" void kernel_launch(void* const* d_in, const int* in_sizes, int n_in,
                              void* d_out, int out_size, void* d_ws, size_t ws_size,
                              hipStream_t stream) {
    (void)in_sizes; (void)n_in; (void)out_size; (void)ws_size;
    const float* x     = (const float*)d_in[0];
    const int*   ei    = (const int*)d_in[1];      // [2,E]: [0]=src(col), [1]=dst(row)
    const float* cc    = (const float*)d_in[2];
    const float* W1L   = (const float*)d_in[3];
    const float* W1H   = (const float*)d_in[4];
    const float* W2L   = (const float*)d_in[5];
    const float* W2H   = (const float*)d_in[6];
    const float* WX    = (const float*)d_in[7];
    const float* lam1  = (const float*)d_in[8];
    const float* lam2  = (const float*)d_in[9];
    const float* lin_w = (const float*)d_in[10];
    const float* lin_b = (const float*)d_in[11];
    float* out = (float*)d_out;

    float* ws    = (float*)d_ws;
    float* degL  = ws;                 // becomes dinv after k_dinv
    float* degLL = degL  + NN;
    float* degH  = degLL + NN;
    float* degHH = degH  + NN;
    float* vl    = ws + 4 * NN;
    float* vll   = vl  + EE;
    float* vh    = vll + EE;
    float* vhh   = vh  + EE;
    float* t0    = vhh + EE;
    float* t1    = t0 + (size_t)NN * HH;
    float* t2    = t1 + (size_t)NN * HH;
    float* t3    = t2 + (size_t)NN * HH;
    unsigned short* b0   = (unsigned short*)(t3 + (size_t)NN * HH);
    unsigned short* b1   = b0 + (size_t)NN * HH;
    unsigned short* wb1L = b1 + (size_t)NN * HH;
    unsigned short* wb1H = wb1L + FF * HH;
    unsigned short* wb2L = wb1H + FF * HH;
    unsigned short* wb2H = wb2L + FF * HH;
    unsigned short* wbX  = wb2H + FF * HH;
    unsigned short* wbLin= wbX  + FF * HH;

    const dim3 b256(256);
    const int nblk_deg = (4 * NN + 255) / 256;
    const int nblk_e   = (EE + 255) / 256;
    const int nblk_nf  = (int)(((size_t)NN * FF + 255) / 256);
    const int nblk_w   = (FF * HH + 255) / 256;
    const int nblk_sc  = (EE + 7) / 8;             // one wave per edge
    const int nblk_gg  = (NN / 16 + 7) / 8;        // 782 blocks, 8 row tiles each

    // one-time weight conversion to bf16
    k_cvt_bf16<<<nblk_w, b256, 0, stream>>>(W1L, wb1L, FF * HH);
    k_cvt_bf16<<<nblk_w, b256, 0, stream>>>(W1H, wb1H, FF * HH);
    k_cvt_bf16<<<nblk_w, b256, 0, stream>>>(W2L, wb2L, FF * HH);
    k_cvt_bf16<<<nblk_w, b256, 0, stream>>>(W2H, wb2H, FF * HH);
    k_cvt_bf16<<<nblk_w, b256, 0, stream>>>(WX,  wbX,  FF * HH);
    k_cvt_bf16<<<(FF * CCL + 255) / 256, b256, 0, stream>>>(lin_w, wbLin, FF * CCL);

    // normalization
    k_init_deg <<<nblk_deg, b256, 0, stream>>>(degL);
    k_edge_deg <<<nblk_e,   b256, 0, stream>>>(ei, cc, degL, degLL, degH, degHH);
    k_dinv     <<<nblk_deg, b256, 0, stream>>>(degL);
    k_edge_vals<<<nblk_e,   b256, 0, stream>>>(ei, cc, degL, degLL, degH, degHH,
                                               vl, vll, vh, vhh);

    // layer 1: y = A_hat x ; x? = relu(y @ W1?)
    k_spmm_diag   <<<nblk_nf, b256, 0, stream>>>(x, degL, t0);
    k_spmm_diag   <<<nblk_nf, b256, 0, stream>>>(x, degH, t1);
    k_spmm_scatter<<<nblk_sc, b256, 0, stream>>>(ei, vl, x, t0);
    k_spmm_scatter<<<nblk_sc, b256, 0, stream>>>(ei, vh, x, t1);
    k_cvt_bf16    <<<nblk_nf, b256, 0, stream>>>(t0, b0, NN * FF);
    k_cvt_bf16    <<<nblk_nf, b256, 0, stream>>>(t1, b1, NN * FF);
    k_gemm<HH, true,  false><<<nblk_gg, b256, 0, stream>>>(b0, wb1L, nullptr, t2, NN);
    k_gemm<HH, true,  false><<<nblk_gg, b256, 0, stream>>>(b1, wb1H, nullptr, t3, NN);

    // layer 2
    k_spmm_diag   <<<nblk_nf, b256, 0, stream>>>(t2, degLL, t0);
    k_spmm_diag   <<<nblk_nf, b256, 0, stream>>>(t3, degHH, t1);
    k_spmm_scatter<<<nblk_sc, b256, 0, stream>>>(ei, vll, t2, t0);
    k_spmm_scatter<<<nblk_sc, b256, 0, stream>>>(ei, vhh, t3, t1);
    k_cvt_bf16    <<<nblk_nf, b256, 0, stream>>>(t0, b0, NN * FF);
    k_cvt_bf16    <<<nblk_nf, b256, 0, stream>>>(t1, b1, NN * FF);
    k_gemm<HH, false, false><<<nblk_gg, b256, 0, stream>>>(b0, wb2L, nullptr, t2, NN);
    k_gemm<HH, false, false><<<nblk_gg, b256, 0, stream>>>(b1, wb2H, nullptr, t3, NN);

    // identity branch + gated fusion + classifier
    k_cvt_bf16<<<nblk_nf, b256, 0, stream>>>(x, b0, NN * FF);
    k_gemm<HH, false, false><<<nblk_gg, b256, 0, stream>>>(b0, wbX, nullptr, t0, NN);
    k_fuse    <<<nblk_nf, b256, 0, stream>>>(t0, t2, t3, cc, lam1, lam2, b1);
    k_gemm<CCL, false, true><<<nblk_gg, b256, 0, stream>>>(b1, wbLin, lin_b, out, NN);
}
